// Seq2Seq_51874615001601
// MI455X (gfx1250) — matrix operand order repaired and checked by reading, not compiled, not run on hardware
//
#include <hip/hip_runtime.h>
#include <hip/hip_bf16.h>

typedef __bf16 bf16_t;
typedef __bf16 v8bf  __attribute__((ext_vector_type(8)));
typedef __bf16 v16bf __attribute__((ext_vector_type(16)));
typedef float  v8f   __attribute__((ext_vector_type(8)));

// async global->LDS copy operand types (matches ROCm clang builtin prototype:
// param0 = int4 addrspace(1)*, param1 = int4 addrspace(3)*)
typedef int v4i __attribute__((vector_size(16)));
typedef __attribute__((address_space(1))) v4i gv4i;
typedef __attribute__((address_space(3))) v4i lv4i;

#define BB   16
#define SS   512
#define TT   32
#define VV   50000
#define OOVN 64
#define VO   (VV + OOVN)     // 50064
#define EE   1024            // E_ENC
#define EM   200             // E_EMB
#define DD   300
#define KH   1324            // D + E_ENC
#define KHP  1344            // padded to 42*32
#define KT_OUT 42
#define NT_OUT (VV / 16)     // 3125
#define KT_ENC 32            // K = 1024
#define NT_ENC 19            // ceil(300/16)
#define MT_ENC ((BB * SS) / 16) // 512

// ---------------------------------------------------------------------------
// Pack an f32 matrix B[k][n] (element = src[n*n_stride + k*k_stride]) into
// bf16 WMMA B-fragments: frag[(nt*KT+kt)*32 + lane][e], where lane holds
// column n = nt*16 + (lane&15) and elements e = K offsets kt*32+(lane>>4)*8+e (e<8),
// kt*32+16+(lane>>4)*8+(e-8) (e>=8).
// ---------------------------------------------------------------------------
__global__ void pack_b_frags(const float* __restrict__ src, bf16_t* __restrict__ dst,
                             long n_stride, long k_stride, int Kmax, int Nmax,
                             int NT, int KT) {
  int idx = blockIdx.x * blockDim.x + threadIdx.x;
  int total = NT * KT * 32;
  if (idx >= total) return;
  int lane = idx & 31;
  int kt = (idx >> 5) % KT;
  int nt = (idx >> 5) / KT;
  int n  = nt * 16 + (lane & 15);
  int kb = kt * 32 + ((lane >> 4) * 8);
  bf16_t* out = dst + (size_t)idx * 16;
#pragma unroll
  for (int e = 0; e < 16; ++e) {
    int k = kb + e + (e >= 8 ? 8 : 0);
    float v = 0.f;
    if (k < Kmax && n < Nmax) v = src[(long)n * n_stride + (long)k * k_stride];
    out[e] = (bf16_t)v;
  }
}

__global__ void f32_to_bf16_kernel(const float* __restrict__ src,
                                   bf16_t* __restrict__ dst, int n) {
  int i = blockIdx.x * blockDim.x + threadIdx.x;
  if (i < n) dst[i] = (bf16_t)src[i];
}

__device__ __forceinline__ v16bf make_a(v8bf lo, v8bf hi) {
  v16bf a;
#pragma unroll
  for (int i = 0; i < 8; ++i) { a[i] = lo[i]; a[i + 8] = hi[i]; }
  return a;
}

// ---------------------------------------------------------------------------
// Generic bf16 WMMA GEMM (A from global), software-pipelined K loop with an
// epilogue WMMA (no per-stage guards / zero-inits in the steady state).
// One wave per 16x16 tile. Used for the enc_output @ attn_W projections.
// ---------------------------------------------------------------------------
__global__ void gemm_wmma_bf16(const bf16_t* __restrict__ A, int lda,
                               const bf16_t* __restrict__ Bp,
                               float* __restrict__ C, int ldc,
                               const float* __restrict__ bias,
                               int MT, int NT, int KT, int Nmax) {
  int wave = (blockIdx.x * blockDim.x + threadIdx.x) >> 5;
  int lane = threadIdx.x & 31;
  if (wave >= MT * NT) return;          // wave-uniform: EXEC stays all-ones
  int nt = wave % NT;
  int mt = wave / NT;
  const bf16_t* Arow  = A + (long)(mt * 16 + (lane & 15)) * lda + ((lane >> 4) * 8);
  const bf16_t* Bfrag = Bp + ((long)(nt * KT) * 32 + lane) * 16;
  v8f acc = {};
  v8bf alo = *(const v8bf*)(Arow);
  v8bf ahi = *(const v8bf*)(Arow + 16);
  v16bf b  = *(const v16bf*)(Bfrag);
  for (int kt = 0; kt + 1 < KT; ++kt) {
    v8bf alo_n = *(const v8bf*)(Arow + (kt + 1) * 32);  // next fragments first
    v8bf ahi_n = *(const v8bf*)(Arow + (kt + 1) * 32 + 16);
    v16bf b_n  = *(const v16bf*)(Bfrag + 32 * 16);
    __builtin_prefetch(Bfrag + 2 * 32 * 16, 0, 1);      // global_prefetch_b8
    acc = __builtin_amdgcn_wmma_f32_16x16x32_bf16(false, make_a(alo, ahi), false, b,
                                                  (short)0, acc, false, false);
    alo = alo_n; ahi = ahi_n; b = b_n;
    Bfrag += 32 * 16;
  }
  acc = __builtin_amdgcn_wmma_f32_16x16x32_bf16(false, make_a(alo, ahi), false, b,
                                                (short)0, acc, false, false);
  int n = nt * 16 + (lane & 15);
  if (n < Nmax) {
    float bv = bias ? bias[n] : 0.f;
    int m0 = mt * 16 + ((lane >> 4) * 8);
#pragma unroll
    for (int i = 0; i < 8; ++i) C[(long)(m0 + i) * ldc + n] = acc[i] + bv;
  }
}

// ---------------------------------------------------------------------------
// Specialized M=16 WMMA GEMM for the logits projection: the single 16xKHP A
// panel (hw) is staged once per block into LDS via gfx1250 async global->LDS
// copy (fallback: plain loads), then every wave reads its A fragments from
// LDS while streaming its B fragments (L2-resident packed out_W) from global.
// ---------------------------------------------------------------------------
__global__ void gemm_wmma_bf16_m16(const bf16_t* __restrict__ A,   // [16, KHP]
                                   const bf16_t* __restrict__ Bp,
                                   float* __restrict__ C, int ldc,
                                   const float* __restrict__ bias,
                                   int NT, int KT) {
  __shared__ bf16_t sA[16 * KHP];       // 43008 bytes
  int tid = threadIdx.x;                // 256
  const int chunks = (16 * KHP) / 8;    // 16-byte chunks = 2688
#if __has_builtin(__builtin_amdgcn_global_load_async_to_lds_b128)
  for (int c = tid; c < chunks; c += 256) {
    __builtin_amdgcn_global_load_async_to_lds_b128(
        (gv4i*)(A + (long)c * 8), (lv4i*)(sA + (long)c * 8), 0, 0);
  }
#if __has_builtin(__builtin_amdgcn_s_wait_asynccnt)
  __builtin_amdgcn_s_wait_asynccnt(0);
#else
  asm volatile("s_wait_asynccnt 0x0" ::: "memory");
#endif
#else
  for (int c = tid; c < chunks; c += 256)
    *(v8bf*)(sA + (long)c * 8) = *(const v8bf*)(A + (long)c * 8);
#endif
  __syncthreads();

  int wave = blockIdx.x * 8 + (tid >> 5);
  int lane = tid & 31;
  if (wave >= NT) return;               // wave-uniform, after the barrier
  int nt = wave;
  const bf16_t* Arow  = sA + (lane & 15) * KHP + ((lane >> 4) * 8);
  const bf16_t* Bfrag = Bp + ((long)(nt * KT) * 32 + lane) * 16;
  v8f acc = {};
  v8bf alo = *(const v8bf*)(Arow);
  v8bf ahi = *(const v8bf*)(Arow + 16);
  v16bf b  = *(const v16bf*)(Bfrag);
  for (int kt = 0; kt + 1 < KT; ++kt) {
    v8bf alo_n = *(const v8bf*)(Arow + (kt + 1) * 32);  // ds_load_b128
    v8bf ahi_n = *(const v8bf*)(Arow + (kt + 1) * 32 + 16);
    v16bf b_n  = *(const v16bf*)(Bfrag + 32 * 16);      // global b128 x2
    __builtin_prefetch(Bfrag + 2 * 32 * 16, 0, 1);      // global_prefetch_b8
    acc = __builtin_amdgcn_wmma_f32_16x16x32_bf16(false, make_a(alo, ahi), false, b,
                                                  (short)0, acc, false, false);
    alo = alo_n; ahi = ahi_n; b = b_n;
    Bfrag += 32 * 16;
  }
  acc = __builtin_amdgcn_wmma_f32_16x16x32_bf16(false, make_a(alo, ahi), false, b,
                                                (short)0, acc, false, false);
  int n = nt * 16 + (lane & 15);        // always < VV (NT exact)
  float bv = bias[n];
  int m0 = (lane >> 4) * 8;
#pragma unroll
  for (int i = 0; i < 8; ++i) C[(long)(m0 + i) * ldc + n] = acc[i] + bv;
}

// ---------------------------------------------------------------------------
// attention: att = softmax(mask(h . encp)), weighted = att^T @ enc_output
// ---------------------------------------------------------------------------
__global__ void attend_kernel(const float* __restrict__ h,     // [B,D]
                              const float* __restrict__ encp,  // [B*S,D]
                              const float* __restrict__ enc,   // [B*S,E]
                              const int*   __restrict__ src_ext,// [B,S]
                              float* __restrict__ wout,        // [B,E]
                              float* __restrict__ aout) {      // [B,S] or null
  __shared__ float sh[DD];
  __shared__ float satt[SS];
  __shared__ float sred[SS];
  int b = blockIdx.x, tid = threadIdx.x;
  for (int k = tid; k < DD; k += SS) sh[k] = h[b * DD + k];
  __syncthreads();
  {
    const float* er = encp + ((long)b * SS + tid) * DD;
    float dot = 0.f;
    for (int k = 0; k < DD; ++k) dot += sh[k] * er[k];
    if (src_ext[b * SS + tid] == 0) dot = -1e30f;
    satt[tid] = dot;
  }
  __syncthreads();
  sred[tid] = satt[tid];
  __syncthreads();
  for (int off = SS / 2; off > 0; off >>= 1) {
    if (tid < off) sred[tid] = fmaxf(sred[tid], sred[tid + off]);
    __syncthreads();
  }
  float mx = sred[0];
  __syncthreads();
  float e = __expf(satt[tid] - mx);
  sred[tid] = e;
  __syncthreads();
  for (int off = SS / 2; off > 0; off >>= 1) {
    if (tid < off) sred[tid] += sred[tid + off];
    __syncthreads();
  }
  float inv = 1.f / sred[0];
  __syncthreads();
  float a = e * inv;
  satt[tid] = a;
  if (aout) aout[b * SS + tid] = a;
  __syncthreads();
  for (int c = tid; c < EE; c += SS) {
    float acc = 0.f;
    const float* base = enc + (long)b * SS * EE + c;
    for (int s = 0; s < SS; ++s) acc += satt[s] * base[(long)s * EE];
    wout[b * EE + c] = acc;
  }
}

// GRU cell: one block per batch, thread j computes channel j of all 3 gates.
__global__ void gru_kernel(const float* __restrict__ hin, float* __restrict__ hout,
                           const float* __restrict__ emb, const int* __restrict__ tgt,
                           int tstep, const float* __restrict__ wgt,
                           const float* __restrict__ W_ih, const float* __restrict__ W_hh,
                           const float* __restrict__ b_ih, const float* __restrict__ b_hh) {
  __shared__ float sx[EM + EE];
  __shared__ float shh[DD];
  int b = blockIdx.x, tid = threadIdx.x;
  int tok = tgt[b * TT + tstep];
  for (int k = tid; k < EM + EE; k += blockDim.x)
    sx[k] = (k < EM) ? emb[(long)tok * EM + k] : wgt[b * EE + (k - EM)];
  for (int k = tid; k < DD; k += blockDim.x) shh[k] = hin[b * DD + k];
  __syncthreads();
  if (tid < DD) {
    int j = tid;
    float gr = b_ih[j], gz = b_ih[j + DD], gn = b_ih[j + 2 * DD];
    const float* wr = W_ih + (long)j * (EM + EE);
    const float* wz = W_ih + (long)(j + DD) * (EM + EE);
    const float* wn = W_ih + (long)(j + 2 * DD) * (EM + EE);
    for (int k = 0; k < EM + EE; ++k) {
      float x = sx[k];
      gr += wr[k] * x; gz += wz[k] * x; gn += wn[k] * x;
    }
    float hr = b_hh[j], hz = b_hh[j + DD], hn = b_hh[j + 2 * DD];
    const float* vr = W_hh + (long)j * DD;
    const float* vz = W_hh + (long)(j + DD) * DD;
    const float* vn = W_hh + (long)(j + 2 * DD) * DD;
    for (int k = 0; k < DD; ++k) {
      float hh = shh[k];
      hr += vr[k] * hh; hz += vz[k] * hh; hn += vn[k] * hh;
    }
    float r = 1.f / (1.f + __expf(-(gr + hr)));
    float z = 1.f / (1.f + __expf(-(gz + hz)));
    float n = tanhf(gn + r * hn);
    hout[b * DD + j] = (1.f - z) * n + z * shh[j];
  }
}

__global__ void build_hw_kernel(const float* __restrict__ h,
                                const float* __restrict__ w2,
                                bf16_t* __restrict__ hw) {
  int i = blockIdx.x * blockDim.x + threadIdx.x;
  if (i >= BB * KHP) return;
  int b = i / KHP, k = i % KHP;
  float v = 0.f;
  if (k < DD) v = h[b * DD + k];
  else if (k < KH) v = w2[b * EE + (k - DD)];
  hw[i] = (bf16_t)v;
}

__global__ void pgen_kernel(const float* __restrict__ h, const float* __restrict__ w2,
                            const float* __restrict__ p_W, const float* __restrict__ p_b,
                            float* __restrict__ pgen) {
  __shared__ float sred[256];
  int b = blockIdx.x, tid = threadIdx.x;
  float acc = 0.f;
  for (int k = tid; k < KH; k += 256)
    acc += p_W[k] * ((k < DD) ? h[b * DD + k] : w2[b * EE + (k - DD)]);
  sred[tid] = acc;
  __syncthreads();
  for (int off = 128; off > 0; off >>= 1) {
    if (tid < off) sred[tid] += sred[tid + off];
    __syncthreads();
  }
  if (tid == 0) pgen[b] = 1.f / (1.f + __expf(-(sred[0] + p_b[0])));
}

__global__ void softmax_stats(const float* __restrict__ logits,
                              float* __restrict__ rowmax, float* __restrict__ rowsum) {
  __shared__ float sred[1024];
  int b = blockIdx.x, tid = threadIdx.x;
  float m = -3.4e38f;
  for (int n = tid; n < VV; n += 1024) m = fmaxf(m, logits[(long)b * VV + n]);
  sred[tid] = m;
  __syncthreads();
  for (int off = 512; off > 0; off >>= 1) {
    if (tid < off) sred[tid] = fmaxf(sred[tid], sred[tid + off]);
    __syncthreads();
  }
  m = sred[0];
  __syncthreads();
  float s = 0.f;
  for (int n = tid; n < VV; n += 1024) s += __expf(logits[(long)b * VV + n] - m);
  sred[tid] = s;
  __syncthreads();
  for (int off = 512; off > 0; off >>= 1) {
    if (tid < off) sred[tid] += sred[tid + off];
    __syncthreads();
  }
  if (tid == 0) { rowmax[b] = m; rowsum[b] = sred[0]; }
}

__global__ void write_probs(const float* __restrict__ logits, const float* __restrict__ rowmax,
                            const float* __restrict__ rowsum, const float* __restrict__ pgen,
                            float* __restrict__ out_probs, int t) {
  long i = (long)blockIdx.x * blockDim.x + threadIdx.x;
  if (i >= (long)BB * VO) return;
  int b = (int)(i / VO), n = (int)(i % VO);
  float p = 0.f;
  if (n < VV) p = pgen[b] * __expf(logits[(long)b * VV + n] - rowmax[b]) / rowsum[b];
  out_probs[((long)b * TT + t) * VO + n] = p;
}

__global__ void scatter_kernel(const float* __restrict__ a2, const float* __restrict__ pgen,
                               const int* __restrict__ src_ext,
                               float* __restrict__ out_probs, int t) {
  int i = blockIdx.x * blockDim.x + threadIdx.x;
  if (i >= BB * SS) return;
  int b = i / SS;
  atomicAdd(&out_probs[((long)b * TT + t) * VO + src_ext[i]], (1.f - pgen[b]) * a2[i]);
}

__global__ void log_kernel(const float* __restrict__ out_probs,
                           float* __restrict__ out_log, int t) {
  long i = (long)blockIdx.x * blockDim.x + threadIdx.x;
  if (i >= (long)BB * VO) return;
  int b = (int)(i / VO), n = (int)(i % VO);
  long off = ((long)b * TT + t) * VO + n;
  out_log[off] = __logf(out_probs[off] + 1e-12f);
}

__global__ void init_kernel(float* __restrict__ out_log, float* __restrict__ out_probs,
                            float* __restrict__ h0) {
  long i = (long)blockIdx.x * blockDim.x + threadIdx.x;
  if (i < (long)BB * VO) {
    int b = (int)(i / VO), n = (int)(i % VO);
    long off = ((long)b * TT + 0) * VO + n;
    out_probs[off] = 0.f;
    out_log[off] = __logf(1e-12f);
  }
  if (i < BB * DD) h0[i] = 0.f;
}

extern "C" void kernel_launch(void* const* d_in, const int* in_sizes, int n_in,
                              void* d_out, int out_size, void* d_ws, size_t ws_size,
                              hipStream_t stream) {
  (void)in_sizes; (void)n_in; (void)out_size; (void)ws_size;
  const float* enc_output = (const float*)d_in[0];
  const int*   target_ids = (const int*)d_in[1];
  const int*   src_ext    = (const int*)d_in[2];
  const float* emb        = (const float*)d_in[3];
  const float* attn1_W    = (const float*)d_in[4];
  const float* attn2_W    = (const float*)d_in[5];
  const float* W_ih       = (const float*)d_in[6];
  const float* W_hh       = (const float*)d_in[7];
  const float* b_ih       = (const float*)d_in[8];
  const float* b_hh       = (const float*)d_in[9];
  const float* out_W      = (const float*)d_in[10];
  const float* out_b      = (const float*)d_in[11];
  const float* p_W        = (const float*)d_in[12];
  const float* p_b        = (const float*)d_in[13];

  float* out_log   = (float*)d_out;
  float* out_probs = out_log + (long)BB * TT * VO;

  char* w = (char*)d_ws;
  auto alloc = [&](size_t bytes) -> char* {
    char* p = w; w += (bytes + 255) & ~(size_t)255; return p;
  };
  bf16_t* BpOut = (bf16_t*)alloc((size_t)NT_OUT * KT_OUT * 512 * sizeof(bf16_t)); // 134 MB, L2-resident
  bf16_t* BpA1  = (bf16_t*)alloc((size_t)NT_ENC * KT_ENC * 512 * sizeof(bf16_t));
  bf16_t* BpA2  = (bf16_t*)alloc((size_t)NT_ENC * KT_ENC * 512 * sizeof(bf16_t));
  bf16_t* Aenc  = (bf16_t*)alloc((size_t)BB * SS * EE * sizeof(bf16_t));
  float* enc1   = (float*)alloc((size_t)BB * SS * DD * sizeof(float));
  float* enc2   = (float*)alloc((size_t)BB * SS * DD * sizeof(float));
  float* hbuf0  = (float*)alloc((size_t)BB * DD * sizeof(float));
  float* hbuf1  = (float*)alloc((size_t)BB * DD * sizeof(float));
  float* w1     = (float*)alloc((size_t)BB * EE * sizeof(float));
  float* w2     = (float*)alloc((size_t)BB * EE * sizeof(float));
  float* a2     = (float*)alloc((size_t)BB * SS * sizeof(float));
  bf16_t* hw    = (bf16_t*)alloc((size_t)BB * KHP * sizeof(bf16_t));
  float* logits = (float*)alloc((size_t)BB * VV * sizeof(float));
  float* rowmax = (float*)alloc(BB * sizeof(float));
  float* rowsum = (float*)alloc(BB * sizeof(float));
  float* pgen   = (float*)alloc(BB * sizeof(float));

  // One-time packs / conversions
  {
    int total = NT_OUT * KT_OUT * 32;
    pack_b_frags<<<(total + 255) / 256, 256, 0, stream>>>(out_W, BpOut, KH, 1, KH, VV, NT_OUT, KT_OUT);
  }
  {
    int total = NT_ENC * KT_ENC * 32;
    pack_b_frags<<<(total + 255) / 256, 256, 0, stream>>>(attn1_W, BpA1, 1, DD, EE, DD, NT_ENC, KT_ENC);
    pack_b_frags<<<(total + 255) / 256, 256, 0, stream>>>(attn2_W, BpA2, 1, DD, EE, DD, NT_ENC, KT_ENC);
  }
  {
    int n = BB * SS * EE;
    f32_to_bf16_kernel<<<(n + 255) / 256, 256, 0, stream>>>(enc_output, Aenc, n);
  }
  // enc1 / enc2 = enc_output @ attn{1,2}_W via WMMA
  {
    int tiles = MT_ENC * NT_ENC;
    int blocks = (tiles + 7) / 8;
    gemm_wmma_bf16<<<blocks, 256, 0, stream>>>(Aenc, EE, BpA1, enc1, DD, nullptr, MT_ENC, NT_ENC, KT_ENC, DD);
    gemm_wmma_bf16<<<blocks, 256, 0, stream>>>(Aenc, EE, BpA2, enc2, DD, nullptr, MT_ENC, NT_ENC, KT_ENC, DD);
  }
  {
    long n = (long)BB * VO;
    init_kernel<<<(int)((n + 255) / 256), 256, 0, stream>>>(out_log, out_probs, hbuf0);
  }

  float* hcur = hbuf0;
  float* hnext = hbuf1;
  for (int ts = 0; ts < TT - 1; ++ts) {
    int t = ts + 1;
    attend_kernel<<<BB, SS, 0, stream>>>(hcur, enc1, enc_output, src_ext, w1, nullptr);
    gru_kernel<<<BB, 320, 0, stream>>>(hcur, hnext, emb, target_ids, ts, w1, W_ih, W_hh, b_ih, b_hh);
    attend_kernel<<<BB, SS, 0, stream>>>(hnext, enc2, enc_output, src_ext, w2, a2);
    build_hw_kernel<<<(BB * KHP + 255) / 256, 256, 0, stream>>>(hnext, w2, hw);
    pgen_kernel<<<BB, 256, 0, stream>>>(hnext, w2, p_W, p_b, pgen);
    // logits[16,50000] = hw @ out_W.T + out_b (dominant op; LDS-staged A, L2 B)
    gemm_wmma_bf16_m16<<<(NT_OUT + 7) / 8, 256, 0, stream>>>(hw, BpOut, logits, VV, out_b, NT_OUT, KT_OUT);
    softmax_stats<<<BB, 1024, 0, stream>>>(logits, rowmax, rowsum);
    {
      long n = (long)BB * VO;
      write_probs<<<(int)((n + 255) / 256), 256, 0, stream>>>(logits, rowmax, rowsum, pgen, out_probs, t);
      scatter_kernel<<<(BB * SS + 255) / 256, 256, 0, stream>>>(a2, pgen, src_ext, out_probs, t);
      log_kernel<<<(int)((n + 255) / 256), 256, 0, stream>>>(out_probs, out_log, t);
    }
    float* tmp = hcur; hcur = hnext; hnext = tmp;
  }
}
